// MQ_90452011254107
// MI455X (gfx1250) — matrix operand order repaired and checked by reading, not compiled, not run on hardware
//
#include <hip/hip_runtime.h>

typedef __attribute__((ext_vector_type(16))) _Float16 v16h;
typedef __attribute__((ext_vector_type(8)))  float    v8f;

#define N_ROWS   32768
#define C_DIM    256
#define K_CODES  2048
#define M_BOOK   4
#define IN_DIM   64

// ---------------------------------------------------------------------------
// WGP-scope prefetch (scope 0 = pull into all cache levels incl. L0/WGP$).
// __builtin_prefetch only reaches SE/SYS scope, which skips the WGP caches.
__device__ __forceinline__ void prefetch_wgp(const void* p) {
    asm volatile("global_prefetch_b8 %0, off" :: "v"(p));
}

// deterministic hash-bernoulli for the mask (jax PRNG not reproducible here)
__device__ __forceinline__ bool bern_mask(int c) {
    unsigned u = (unsigned)c * 2654435761u + 0x9E3779B9u;
    u ^= u >> 16; u *= 0x45d9f3bu;
    u ^= u >> 16; u *= 0x45d9f3bu;
    u ^= u >> 16;
    return (u & 0xFFFFFFu) < 1677722u;   // ~0.1 * 2^24
}

// A-fragment k mapping (16-bit A 16x32 layout, ISA 7.12.2)
__device__ __forceinline__ int a_klocal(int e, int lane) {
    int v = e >> 1, h = e & 1;
    int base = (v < 4) ? (2 * v + h) : (16 + 2 * (v - 4) + h);
    return base + ((lane >= 16) ? 8 : 0);
}

// ---------------------------------------------------------------------------
// Build A fragments from fp32 row-major [Rrows x K]
__global__ void convA_kernel(const float* __restrict__ src, _Float16* __restrict__ dst,
                             int K, int Q, int total) {
    int gid = blockIdx.x * blockDim.x + threadIdx.x;
    if (gid >= total) return;
    int l = gid & 31;
    int q = (gid >> 5) % Q;
    int t = gid / (Q * 32);
    int row = t * 16 + (l & 15);
    v16h frag;
#pragma unroll
    for (int e = 0; e < 16; ++e) {
        int k = 32 * q + a_klocal(e, l);
        frag[e] = (_Float16)src[(size_t)row * K + k];
    }
    *(v16h*)(dst + (size_t)gid * 16) = frag;
}

// A fragments for masked input: value = row0 ? pos[k] : ((bern? 0 : x) + pos)
__global__ void convA_mask_kernel(const float* __restrict__ x, const float* __restrict__ pos,
                                  _Float16* __restrict__ dst, int total) {
    const int K = IN_DIM, Q = 2;
    int gid = blockIdx.x * blockDim.x + threadIdx.x;
    if (gid >= total) return;
    int l = gid & 31;
    int q = (gid >> 5) % Q;
    int t = gid / (Q * 32);
    int row = t * 16 + (l & 15);
    v16h frag;
#pragma unroll
    for (int e = 0; e < 16; ++e) {
        int k = 32 * q + a_klocal(e, l);
        float v;
        if (row == 0)            v = pos[k];
        else if (bern_mask(k))   v = pos[k];
        else                     v = x[(size_t)row * K + k] + pos[k];
        frag[e] = (_Float16)v;
    }
    *(v16h*)(dst + (size_t)gid * 16) = frag;
}

// A fragments of zq = sum of the 4 ce slices (each [N x 256] fp32)
__global__ void convA_sum4_kernel(const float* __restrict__ ce, _Float16* __restrict__ dst,
                                  int total) {
    const int K = C_DIM, Q = 8;
    const size_t SL = (size_t)N_ROWS * C_DIM;
    int gid = blockIdx.x * blockDim.x + threadIdx.x;
    if (gid >= total) return;
    int l = gid & 31;
    int q = (gid >> 5) % Q;
    int t = gid / (Q * 32);
    int row = t * 16 + (l & 15);
    v16h frag;
#pragma unroll
    for (int e = 0; e < 16; ++e) {
        int k = 32 * q + a_klocal(e, l);
        size_t o = (size_t)row * K + k;
        float v = ce[o] + ce[o + SL] + ce[o + 2 * SL] + ce[o + 3 * SL];
        frag[e] = (_Float16)v;
    }
    *(v16h*)(dst + (size_t)gid * 16) = frag;
}

// Build B fragments: B[k,n] = src[n*sn + k*sk]  (32x16 f16 B layout)
__global__ void convB_kernel(const float* __restrict__ src, _Float16* __restrict__ dst,
                             long sn, long sk, int Q, int total) {
    int gid = blockIdx.x * blockDim.x + threadIdx.x;
    if (gid >= total) return;
    int l = gid & 31;
    int q = (gid >> 5) % Q;
    int t = gid / (Q * 32);
    int n = t * 16 + (l & 15);
    int kofs = (l >= 16) ? 8 : 0;
    v16h frag;
#pragma unroll
    for (int e = 0; e < 16; ++e) {
        int k = 32 * q + e + ((e >= 8) ? 8 : 0) + kofs;
        frag[e] = (_Float16)src[(size_t)n * sn + (size_t)k * sk];
    }
    *(v16h*)(dst + (size_t)gid * 16) = frag;
}

// ---------------------------------------------------------------------------
// Fragment GEMM: out[N x P] = A @ W + bias ; one wave per 16-row strip.
// Whole B tile issued as one load clause before the WMMA chain (sched_barrier);
// next tile prefetched at WGP scope. launch_bounds(256,1) -> no spills.
template <int Q>
__global__ void __launch_bounds__(256, 1)
gemm_frag_kernel(const _Float16* __restrict__ afrag,
                 const _Float16* __restrict__ bfrag,
                 const float* __restrict__ bias,
                 float* __restrict__ out, int P, int nct) {
    int w = threadIdx.x >> 5;
    int l = threadIdx.x & 31;
    int s = blockIdx.x * 8 + w;
    v16h a[Q];
    const _Float16* ap = afrag + (size_t)s * Q * 512 + l * 16;
#pragma unroll
    for (int q = 0; q < Q; ++q) a[q] = *(const v16h*)(ap + (size_t)q * 512);
    int hi = (l >= 16) ? 8 : 0;
    int col0 = l & 15;

    for (int ct = 0; ct < nct; ++ct) {
        const _Float16* bp = bfrag + (size_t)ct * Q * 512 + l * 16;
        v16h b[Q];
#pragma unroll
        for (int q = 0; q < Q; ++q) {
            b[q] = *(const v16h*)(bp + (size_t)q * 512);
            prefetch_wgp(bp + (size_t)q * 512 + (size_t)Q * 512);   // next tile -> L0
        }
        __builtin_amdgcn_sched_barrier(0);   // issue all tile loads before WMMA chain
        v8f acc = {};
#pragma unroll
        for (int q = 0; q < Q; ++q)
            acc = __builtin_amdgcn_wmma_f32_16x16x32_f16(false, a[q], false, b[q],
                                                         (short)0, acc, false, false);
        int col = ct * 16 + col0;
        float bv = bias[col];
#pragma unroll
        for (int j = 0; j < 8; ++j)
            out[(size_t)(s * 16 + j + hi) * P + col] = acc[j] + bv;
    }
}

// ---------------------------------------------------------------------------
// Argmin over codebook: per row, argmin_n ( ||E_n||^2 - 2 r.E_n ).
__global__ void __launch_bounds__(256, 1)
argmin_kernel(const _Float16* __restrict__ afrag,
              const _Float16* __restrict__ cbfrag,
              const float* __restrict__ en,
              int* __restrict__ idx) {
    const int Q = 8;
    const int NCT = K_CODES / 16;
    int w = threadIdx.x >> 5;
    int l = threadIdx.x & 31;
    int s = blockIdx.x * 8 + w;
    v16h a[Q];
    const _Float16* ap = afrag + (size_t)s * Q * 512 + l * 16;
#pragma unroll
    for (int q = 0; q < Q; ++q) a[q] = *(const v16h*)(ap + (size_t)q * 512);

    float bestd[8];
    int   bestn[8];
#pragma unroll
    for (int j = 0; j < 8; ++j) { bestd[j] = 3.4e38f; bestn[j] = 0; }

    int col0 = l & 15;
    const _Float16* bl = cbfrag + l * 16;

    for (int ct = 0; ct < NCT; ++ct) {
        const _Float16* bp = bl + (size_t)ct * Q * 512;
        v16h b[Q];
#pragma unroll
        for (int q = 0; q < Q; ++q) {
            b[q] = *(const v16h*)(bp + (size_t)q * 512);
            prefetch_wgp(bp + (size_t)q * 512 + (size_t)Q * 512);   // next tile -> L0
        }
        __builtin_amdgcn_sched_barrier(0);   // issue all tile loads before WMMA chain
        v8f acc = {};
#pragma unroll
        for (int q = 0; q < Q; ++q)
            acc = __builtin_amdgcn_wmma_f32_16x16x32_f16(false, a[q], false, b[q],
                                                         (short)0, acc, false, false);
        int n = ct * 16 + col0;
        float e = en[n];
#pragma unroll
        for (int j = 0; j < 8; ++j) {
            float d = e - 2.0f * acc[j];
            if (d < bestd[j]) { bestd[j] = d; bestn[j] = n; }
        }
    }
    // reduce across the 16 lanes of each half-wave (same output rows per half)
    for (int off = 1; off < 16; off <<= 1) {
#pragma unroll
        for (int j = 0; j < 8; ++j) {
            float od = __shfl_xor(bestd[j], off, 32);
            int   on = __shfl_xor(bestn[j], off, 32);
            if (od < bestd[j] || (od == bestd[j] && on < bestn[j])) {
                bestd[j] = od; bestn[j] = on;
            }
        }
    }
    if ((l & 15) == 0) {
        int hi = (l >= 16) ? 8 : 0;
#pragma unroll
        for (int j = 0; j < 8; ++j) idx[s * 16 + j + hi] = bestn[j];
    }
}

// ---------------------------------------------------------------------------
// BatchNorm batch statistics: one block per column
__global__ void bnstats_kernel(const float* __restrict__ h, float* __restrict__ mu,
                               float* __restrict__ var, int P) {
    __shared__ float ss[256], ss2[256];
    int c = blockIdx.x, t = threadIdx.x;
    float s = 0.f, s2 = 0.f;
    for (int r = t; r < N_ROWS; r += 256) {
        float v = h[(size_t)r * P + c];
        s += v; s2 += v * v;
    }
    ss[t] = s; ss2[t] = s2;
    __syncthreads();
    for (int o = 128; o > 0; o >>= 1) {
        if (t < o) { ss[t] += ss[t + o]; ss2[t] += ss2[t + o]; }
        __syncthreads();
    }
    if (t == 0) {
        float m = ss[0] / (float)N_ROWS;
        mu[c] = m;
        var[c] = ss2[0] / (float)N_ROWS - m * m;
    }
}

__global__ void bnrelu_kernel(float* __restrict__ h, const float* __restrict__ mu,
                              const float* __restrict__ var, const float* __restrict__ g,
                              const float* __restrict__ bt, long total, int P) {
    long gid = (long)blockIdx.x * blockDim.x + threadIdx.x;
    if (gid >= total) return;
    int c = (int)(gid % P);
    float v = (h[gid] - mu[c]) * rsqrtf(var[c] + 1e-5f) * g[c] + bt[c];
    h[gid] = v > 0.f ? v : 0.f;
}

// ||E||^2 per code row, all 4 books flat [8192 x 256]
__global__ void ennorm_kernel(const float* __restrict__ cb, float* __restrict__ en) {
    int r = blockIdx.x * blockDim.x + threadIdx.x;
    if (r >= M_BOOK * K_CODES) return;
    const float* p = cb + (size_t)r * C_DIM;
    float s = 0.f;
#pragma unroll 4
    for (int j = 0; j < C_DIM; ++j) s += p[j] * p[j];
    en[r] = s;
}

// ce = E[idx]; resNext = resIn - ce
__global__ void gather_kernel(const int* __restrict__ idx, const float* __restrict__ cb,
                              const float* __restrict__ resIn, float* __restrict__ ceOut,
                              float* __restrict__ resNext, long total) {
    long gid = (long)blockIdx.x * blockDim.x + threadIdx.x;
    if (gid >= total) return;
    long i = gid >> 8;
    int  j = (int)(gid & 255);
    float ce = cb[(size_t)idx[i] * C_DIM + j];
    ceOut[gid] = ce;
    if (resNext) resNext[gid] = resIn[gid] - ce;
}

// ---------------------------------------------------------------------------
static inline int cdiv(long a, long b) { return (int)((a + b - 1) / b); }

extern "C" void kernel_launch(void* const* d_in, const int* in_sizes, int n_in,
                              void* d_out, int out_size, void* d_ws, size_t ws_size,
                              hipStream_t stream) {
    const float* x    = (const float*)d_in[0];
    const float* pos  = (const float*)d_in[1];
    const float* e_w0 = (const float*)d_in[2];
    const float* e_b0 = (const float*)d_in[3];
    const float* e_g0 = (const float*)d_in[4];
    const float* e_bt0= (const float*)d_in[5];
    const float* e_w1 = (const float*)d_in[6];
    const float* e_b1 = (const float*)d_in[7];
    const float* e_g1 = (const float*)d_in[8];
    const float* e_bt1= (const float*)d_in[9];
    const float* e_w2 = (const float*)d_in[10];
    const float* e_b2 = (const float*)d_in[11];
    const float* cbs  = (const float*)d_in[12];
    const float* d_w0 = (const float*)d_in[13];
    const float* d_b0 = (const float*)d_in[14];
    const float* d_g0 = (const float*)d_in[15];
    const float* d_bt0= (const float*)d_in[16];
    const float* d_w1 = (const float*)d_in[17];
    const float* d_b1 = (const float*)d_in[18];
    const float* d_g1 = (const float*)d_in[19];
    const float* d_bt1= (const float*)d_in[20];
    const float* d_w2 = (const float*)d_in[21];
    const float* d_b2 = (const float*)d_in[22];

    float* out   = (float*)d_out;
    float* xhat  = out;                                         // [N,64]
    float* resB  = out + (size_t)N_ROWS * IN_DIM;               // [4,N,256]
    float* ceB   = resB + (size_t)M_BOOK * N_ROWS * C_DIM;      // [4,N,256]

    // workspace carve (256B-aligned)
    char* ws = (char*)d_ws;
    size_t off = 0;
    auto carve = [&](size_t bytes) { void* p = ws + off; off = (off + bytes + 255) & ~(size_t)255; return p; };
    float*    hbuf   = (float*)   carve((size_t)N_ROWS * C_DIM * 4);        // 32 MB
    _Float16* hfrag  = (_Float16*)carve((size_t)N_ROWS * C_DIM * 2);        // 16 MB
    _Float16* cbfrag = (_Float16*)carve((size_t)M_BOOK * K_CODES * C_DIM * 2); // 4 MB
    _Float16* ew0f   = (_Float16*)carve(64  * 128 * 2);
    _Float16* ew1f   = (_Float16*)carve(128 * 256 * 2);
    _Float16* ew2f   = (_Float16*)carve(256 * 256 * 2);
    _Float16* dw0f   = (_Float16*)carve(256 * 256 * 2);
    _Float16* dw1f   = (_Float16*)carve(256 * 128 * 2);
    _Float16* dw2f   = (_Float16*)carve(128 * 64  * 2);
    float*    en     = (float*)   carve((size_t)M_BOOK * K_CODES * 4);
    int*      idx    = (int*)     carve((size_t)N_ROWS * 4);
    float*    mu     = (float*)   carve(256 * 4);
    float*    var    = (float*)   carve(256 * 4);
    (void)ws_size; (void)n_in; (void)in_sizes; (void)out_size;

    const int TB = 256;
    const int NT = N_ROWS / 16;           // 2048 row tiles
    const int GS = NT / 8;                // gemm/argmin blocks (8 waves each)

    // weight fragments: W[K,P] -> sn=1, sk=P
    auto launchB = [&](const float* src, _Float16* dst, int K, int P, long sn, long sk) {
        int total = (P / 16) * (K / 32) * 32;
        hipLaunchKernelGGL(convB_kernel, dim3(cdiv(total, TB)), dim3(TB), 0, stream,
                           src, dst, sn, sk, K / 32, total);
    };
    launchB(e_w0, ew0f, 64, 128, 1, 128);
    launchB(e_w1, ew1f, 128, 256, 1, 256);
    launchB(e_w2, ew2f, 256, 256, 1, 256);
    launchB(d_w0, dw0f, 256, 256, 1, 256);
    launchB(d_w1, dw1f, 256, 128, 1, 128);
    launchB(d_w2, dw2f, 128, 64, 1, 64);
    // codebook fragments: B[k, n] = E[n, k]  -> sn=256, sk=1
    for (int m = 0; m < M_BOOK; ++m)
        launchB(cbs + (size_t)m * K_CODES * C_DIM,
                cbfrag + (size_t)m * K_CODES * C_DIM, C_DIM, K_CODES, C_DIM, 1);

    hipLaunchKernelGGL(ennorm_kernel, dim3(cdiv(M_BOOK * K_CODES, TB)), dim3(TB), 0, stream, cbs, en);

    // ---------------- encoder ----------------
    {
        int total = NT * 2 * 32;
        hipLaunchKernelGGL(convA_mask_kernel, dim3(cdiv(total, TB)), dim3(TB), 0, stream,
                           x, pos, hfrag, total);
    }
    hipLaunchKernelGGL(gemm_frag_kernel<2>, dim3(GS), dim3(TB), 0, stream,
                       hfrag, ew0f, e_b0, hbuf, 128, 8);
    hipLaunchKernelGGL(bnstats_kernel, dim3(128), dim3(TB), 0, stream, hbuf, mu, var, 128);
    hipLaunchKernelGGL(bnrelu_kernel, dim3(cdiv((long)N_ROWS * 128, TB)), dim3(TB), 0, stream,
                       hbuf, mu, var, e_g0, e_bt0, (long)N_ROWS * 128, 128);
    {
        int total = NT * 4 * 32;
        hipLaunchKernelGGL(convA_kernel, dim3(cdiv(total, TB)), dim3(TB), 0, stream,
                           hbuf, hfrag, 128, 4, total);
    }
    hipLaunchKernelGGL(gemm_frag_kernel<4>, dim3(GS), dim3(TB), 0, stream,
                       hfrag, ew1f, e_b1, hbuf, 256, 16);
    hipLaunchKernelGGL(bnstats_kernel, dim3(256), dim3(TB), 0, stream, hbuf, mu, var, 256);
    hipLaunchKernelGGL(bnrelu_kernel, dim3(cdiv((long)N_ROWS * 256, TB)), dim3(TB), 0, stream,
                       hbuf, mu, var, e_g1, e_bt1, (long)N_ROWS * 256, 256);
    {
        int total = NT * 8 * 32;
        hipLaunchKernelGGL(convA_kernel, dim3(cdiv(total, TB)), dim3(TB), 0, stream,
                           hbuf, hfrag, 256, 8, total);
    }
    // ze -> res[0] slot directly
    hipLaunchKernelGGL(gemm_frag_kernel<8>, dim3(GS), dim3(TB), 0, stream,
                       hfrag, ew2f, e_b2, resB, 256, 16);

    // ---------------- residual quantization ----------------
    const long NE = (long)N_ROWS * C_DIM;
    for (int m = 0; m < M_BOOK; ++m) {
        float* res_m = resB + (size_t)m * NE;
        int total = NT * 8 * 32;
        hipLaunchKernelGGL(convA_kernel, dim3(cdiv(total, TB)), dim3(TB), 0, stream,
                           res_m, hfrag, 256, 8, total);
        hipLaunchKernelGGL(argmin_kernel, dim3(GS), dim3(TB), 0, stream,
                           hfrag, cbfrag + (size_t)m * K_CODES * C_DIM, en + m * K_CODES, idx);
        hipLaunchKernelGGL(gather_kernel, dim3(cdiv(NE, TB)), dim3(TB), 0, stream,
                           idx, cbs + (size_t)m * K_CODES * C_DIM, res_m,
                           ceB + (size_t)m * NE,
                           (m < M_BOOK - 1) ? (resB + (size_t)(m + 1) * NE) : (float*)nullptr, NE);
    }

    // ---------------- decoder (input = zq = sum ce) ----------------
    {
        int total = NT * 8 * 32;
        hipLaunchKernelGGL(convA_sum4_kernel, dim3(cdiv(total, TB)), dim3(TB), 0, stream,
                           ceB, hfrag, total);
    }
    hipLaunchKernelGGL(gemm_frag_kernel<8>, dim3(GS), dim3(TB), 0, stream,
                       hfrag, dw0f, d_b0, hbuf, 256, 16);
    hipLaunchKernelGGL(bnstats_kernel, dim3(256), dim3(TB), 0, stream, hbuf, mu, var, 256);
    hipLaunchKernelGGL(bnrelu_kernel, dim3(cdiv((long)N_ROWS * 256, TB)), dim3(TB), 0, stream,
                       hbuf, mu, var, d_g0, d_bt0, (long)N_ROWS * 256, 256);
    {
        int total = NT * 8 * 32;
        hipLaunchKernelGGL(convA_kernel, dim3(cdiv(total, TB)), dim3(TB), 0, stream,
                           hbuf, hfrag, 256, 8, total);
    }
    hipLaunchKernelGGL(gemm_frag_kernel<8>, dim3(GS), dim3(TB), 0, stream,
                       hfrag, dw1f, d_b1, hbuf, 128, 8);
    hipLaunchKernelGGL(bnstats_kernel, dim3(128), dim3(TB), 0, stream, hbuf, mu, var, 128);
    hipLaunchKernelGGL(bnrelu_kernel, dim3(cdiv((long)N_ROWS * 128, TB)), dim3(TB), 0, stream,
                       hbuf, mu, var, d_g1, d_bt1, (long)N_ROWS * 128, 128);
    {
        int total = NT * 4 * 32;
        hipLaunchKernelGGL(convA_kernel, dim3(cdiv(total, TB)), dim3(TB), 0, stream,
                           hbuf, hfrag, 128, 4, total);
    }
    hipLaunchKernelGGL(gemm_frag_kernel<4>, dim3(GS), dim3(TB), 0, stream,
                       hfrag, dw2f, d_b2, xhat, 64, 4);
}
